// senet_72971494359214
// MI455X (gfx1250) — matrix operand order repaired and checked by reading, not compiled, not run on hardware
//
#include <hip/hip_runtime.h>
#include <hip/hip_bf16.h>
#include <math.h>

// ---------------------------------------------------------------------------
// SE-Net block for MI455X (gfx1250, wave32, WMMA).
//   pass1: conv1(3x3,4->32)+bias+ReLU, reduce per-(n,oc) sums  (WMMA f16)
//   pass2: SE MLP: mean -> 1x1 relu -> 1x1 sigmoid -> scale[n][32]
//   pass3: recompute conv1 (cheaper than re-reading 512MB h), scale,
//          conv2(3x3,32->4)+bias+ReLU, multiply by x, store     (WMMA f16)
// Convs are implicit GEMMs on v_wmma_f32_16x16x32_f16 (f32 accumulate).
// ---------------------------------------------------------------------------

typedef __attribute__((ext_vector_type(16))) _Float16 v16h;
typedef __attribute__((ext_vector_type(8)))  float    v8f;

union Frag16 { v16h v; unsigned int u[8]; };

__device__ __forceinline__ unsigned int pack2f16(float lo, float hi) {
  union { unsigned int u; _Float16 h[2]; } p;
  p.h[0] = (_Float16)lo; p.h[1] = (_Float16)hi;
  return p.u;
}

__device__ __forceinline__ v8f wmma_f16(const Frag16& a, const Frag16& b, v8f c) {
  // D = A(16x32 f16) * B(32x16 f16) + C(16x16 f32)
  return __builtin_amdgcn_wmma_f32_16x16x32_f16(
      /*neg_a=*/false, a.v, /*neg_b=*/false, b.v,
      /*c_mod=*/(short)0, c, /*reuse_a=*/false, /*reuse_b=*/false);
}

// conv1 B-fragment expansion: frag index f = kt*2+nt, entry e = f*256+lane*8+j.
// B k order used here (same as A gather): VGPR j holds rows K = g*16+2j, +1
// (packed lo/hi), col = lane&15, where g = lane>>4.
__device__ __forceinline__ unsigned int build_w1_entry(const float* __restrict__ w1, int e) {
  int f = e >> 8, w = e & 255, lane = w >> 3, j = w & 7;
  int kt = f >> 1, nt = f & 1;
  int gg = lane >> 4, ncol = lane & 15;
  int K   = kt * 32 + gg * 16 + 2 * j;   // even K: (tap, ch in {0,2})
  int tap = K >> 2, chlo = K & 3;
  int oc  = nt * 16 + ncol;
  float lo = 0.f, hi = 0.f;
  if (tap < 9) {                         // K >= 36 padded with zero
    lo = w1[(oc * 4 + chlo)     * 9 + tap];  // w1[oc][ch][kh][kw], tap=kh*3+kw
    hi = w1[(oc * 4 + chlo + 1) * 9 + tap];
  }
  return pack2f16(lo, hi);
}

// conv1 A-fragment gather from LDS x-patch (f16, [pr][pc][4ch]), pairs as u32.
// A k order used here (same as B): lane group g=ln>>4; pair index t = j + g*8.
template <int PW>
__device__ __forceinline__ void build_a_conv1(const unsigned int* xh32,
                                              int r, int c, int g, int kt, Frag16& a) {
#pragma unroll
  for (int j = 0; j < 8; ++j) {
    int t   = j + g * 8;                 // pair index 0..15 within k-tile
    int tap = kt * 8 + (t >> 1);         // K = kt*32 + 2t = 4*tap + 2*bb
    int bb  = t & 1;                     // channel pair select
    a.u[j]  = (tap < 9)
                ? xh32[((r + tap / 3) * PW + (c + tap % 3)) * 2 + bb]
                : 0u;
  }
}

// ---------------------------------------------------------------------------
__global__ void k_zero(float* __restrict__ p, int n) {
  int i = threadIdx.x + blockIdx.x * blockDim.x;
  if (i < n) p[i] = 0.f;
}

// ---------------------------------------------------------------------------
// Pass 1: 32x32 tile of conv1+ReLU, accumulate per-(n,oc) sums. Grid (16,16,16).
__global__ __launch_bounds__(256) void k_conv1_sum(
    const float* __restrict__ x, const float* __restrict__ w1,
    const float* __restrict__ b1, float* __restrict__ g_sums) {
  const int tid = threadIdx.x;
  const int ln  = tid & 31;
  const int wv  = tid >> 5;
  const int g   = (ln >> 4) & 1;
  const int l15 = ln & 15;               // A: pixel row; D: output column
  const int n   = blockIdx.z;
  const int r0  = blockIdx.y * 32, c0 = blockIdx.x * 32;

  __shared__ __align__(16) _Float16     sh_xh[34 * 34 * 4];
  __shared__ __align__(16) unsigned int sh_w1[4 * 32 * 8];
  __shared__ float sh_sum[32];

  if (tid < 32) sh_sum[tid] = 0.f;

  // Stage x patch (34x34x4) with zero padding, converted to f16.
  for (int pos = tid; pos < 34 * 34; pos += 256) {
    int pr = pos / 34, pc = pos % 34;
    int gr = r0 - 1 + pr, gc = c0 - 1 + pc;
    bool ok = (gr >= 0) & (gr < 512) & (gc >= 0) & (gc < 512);
#pragma unroll
    for (int ch = 0; ch < 4; ++ch) {
      float f = ok ? x[(((size_t)n * 4 + ch) * 512 + gr) * 512 + gc] : 0.f;
      sh_xh[pos * 4 + ch] = (_Float16)f;
    }
  }
  for (int e = tid; e < 4 * 256; e += 256) sh_w1[e] = build_w1_entry(w1, e);
  __syncthreads();

  const unsigned int* xh32 = (const unsigned int*)sh_xh;
  const float bias0 = b1[l15], bias1 = b1[16 + l15];
  float sum0 = 0.f, sum1 = 0.f;

  for (int mt = wv; mt < 64; mt += 8) {       // 1024 pixels / 16 rows
    int p = mt * 16 + l15;
    int r = p >> 5, c = p & 31;
    v8f acc0, acc1;
#pragma unroll
    for (int i = 0; i < 8; ++i) { acc0[i] = bias0; acc1[i] = bias1; }
#pragma unroll
    for (int kt = 0; kt < 2; ++kt) {
      Frag16 a;  build_a_conv1<34>(xh32, r, c, g, kt, a);
      Frag16 bf0, bf1;
#pragma unroll
      for (int j = 0; j < 8; ++j) {
        bf0.u[j] = sh_w1[(kt * 2 + 0) * 256 + ln * 8 + j];
        bf1.u[j] = sh_w1[(kt * 2 + 1) * 256 + ln * 8 + j];
      }
      acc0 = wmma_f16(a, bf0, acc0);
      acc1 = wmma_f16(a, bf1, acc1);
    }
#pragma unroll
    for (int i = 0; i < 8; ++i) {
      sum0 += fmaxf(acc0[i], 0.f);
      sum1 += fmaxf(acc1[i], 0.f);
    }
  }
  atomicAdd(&sh_sum[l15],      sum0);
  atomicAdd(&sh_sum[16 + l15], sum1);
  __syncthreads();
  if (tid < 32) atomicAdd(&g_sums[n * 32 + tid], sh_sum[tid]);
}

// ---------------------------------------------------------------------------
// Pass 2: SE MLP. Grid (16), block 32 (one wave per batch image).
__global__ void k_se(const float* __restrict__ g_sums,
                     const float* __restrict__ w_se1, const float* __restrict__ b_se1,
                     const float* __restrict__ w_se2, const float* __restrict__ b_se2,
                     float* __restrict__ g_scale) {
  const int n = blockIdx.x;
  const int c = threadIdx.x;                  // 0..31
  __shared__ float mean[32];
  __shared__ float q[16];
  mean[c] = g_sums[n * 32 + c] * (1.f / (512.f * 512.f));
  __syncthreads();
  if (c < 16) {
    float a = b_se1[c];
    for (int i = 0; i < 32; ++i) a += w_se1[c * 32 + i] * mean[i];
    q[c] = fmaxf(a, 0.f);
  }
  __syncthreads();
  float t = b_se2[c];
  for (int j = 0; j < 16; ++j) t += w_se2[c * 16 + j] * q[j];
  g_scale[n * 32 + c] = 1.f / (1.f + expf(-t));
}

// ---------------------------------------------------------------------------
// Pass 3: fused recompute-conv1 -> scale -> conv2 -> ReLU -> *x.
// 16x16 output tile; h computed on 18x18 halo (324 px), x patch 20x20.
// Grid (32,32,16).
__global__ __launch_bounds__(256) void k_fused(
    const float* __restrict__ x,
    const float* __restrict__ w1, const float* __restrict__ b1,
    const float* __restrict__ w2, const float* __restrict__ b2,
    const float* __restrict__ g_scale, float* __restrict__ out) {
  const int tid = threadIdx.x;
  const int ln  = tid & 31;
  const int wv  = tid >> 5;
  const int g   = (ln >> 4) & 1;
  const int l15 = ln & 15;
  const int n   = blockIdx.z;
  const int r0  = blockIdx.y * 16, c0 = blockIdx.x * 16;

  __shared__ __align__(16) _Float16     sh_xh[20 * 20 * 4];   //  3.2 KB
  __shared__ __align__(16) _Float16     sh_h[336 * 32];       // 21.5 KB (324 used)
  __shared__ __align__(16) unsigned int sh_w1[4 * 32 * 8];    //  4.0 KB
  __shared__ __align__(16) unsigned int sh_w2[9 * 32 * 8];    //  9.2 KB
  __shared__ float sh_s[32];

  if (tid < 32) sh_s[tid] = g_scale[n * 32 + tid];

  // x patch 20x20x4 -> f16 (zero-padded)
  for (int pos = tid; pos < 20 * 20; pos += 256) {
    int pr = pos / 20, pc = pos % 20;
    int gr = r0 - 2 + pr, gc = c0 - 2 + pc;
    bool ok = (gr >= 0) & (gr < 512) & (gc >= 0) & (gc < 512);
#pragma unroll
    for (int ch = 0; ch < 4; ++ch) {
      float f = ok ? x[(((size_t)n * 4 + ch) * 512 + gr) * 512 + gc] : 0.f;
      sh_xh[pos * 4 + ch] = (_Float16)f;
    }
  }
  // Weight fragments for both convs (once per block).
  for (int e = tid; e < 4 * 256 + 9 * 256; e += 256) {
    if (e < 1024) {
      sh_w1[e] = build_w1_entry(w1, e);
    } else {
      int e2 = e - 1024;
      int tap = e2 >> 8, w = e2 & 255, lane = w >> 3, j = w & 7;
      int gg = lane >> 4, ncol = lane & 15;
      int chlo = gg * 16 + 2 * j;             // conv2 k-tile = one tap, K = channel
      float lo = 0.f, hi = 0.f;
      if (ncol < 4) {                          // only 4 real output channels
        lo = w2[(ncol * 32 + chlo)     * 9 + tap];
        hi = w2[(ncol * 32 + chlo + 1) * 9 + tap];
      }
      sh_w2[e2] = pack2f16(lo, hi);
    }
  }
  __syncthreads();

  // --- conv1 + ReLU + SE scale -> sh_h (f16, [pix 18x18][32ch]) ---
  {
    const unsigned int* xh32 = (const unsigned int*)sh_xh;
    const float bias0 = b1[l15], bias1 = b1[16 + l15];
    const float s0 = sh_s[l15], s1 = sh_s[16 + l15];
    for (int mt = wv; mt < 21; mt += 8) {     // 324 px -> 21 m-tiles (tail clamped)
      int p = mt * 16 + l15;  if (p > 323) p = 323;
      int r = p / 18, c = p % 18;
      v8f acc0, acc1;
#pragma unroll
      for (int i = 0; i < 8; ++i) { acc0[i] = bias0; acc1[i] = bias1; }
#pragma unroll
      for (int kt = 0; kt < 2; ++kt) {
        Frag16 a;  build_a_conv1<20>(xh32, r, c, g, kt, a);
        Frag16 bf0, bf1;
#pragma unroll
        for (int j = 0; j < 8; ++j) {
          bf0.u[j] = sh_w1[(kt * 2 + 0) * 256 + ln * 8 + j];
          bf1.u[j] = sh_w1[(kt * 2 + 1) * 256 + ln * 8 + j];
        }
        acc0 = wmma_f16(a, bf0, acc0);
        acc1 = wmma_f16(a, bf1, acc1);
      }
#pragma unroll
      for (int i = 0; i < 8; ++i) {           // D: lane=col oc, VGPR i = row M
        int pix = mt * 16 + i + g * 8;
        if (pix < 324) {
          sh_h[pix * 32 + l15]      = (_Float16)(fmaxf(acc0[i], 0.f) * s0);
          sh_h[pix * 32 + 16 + l15] = (_Float16)(fmaxf(acc1[i], 0.f) * s1);
        }
      }
    }
  }
  __syncthreads();

  // --- conv2 + bias + ReLU, residual multiply, store ---
  {
    const unsigned int* h32 = (const unsigned int*)sh_h;   // [pix][16 u32 pairs]
    const float bias = (l15 < 4) ? b2[l15] : 0.f;
    for (int mt = wv; mt < 16; mt += 8) {     // 256 out px -> 16 m-tiles
      int p = mt * 16 + l15;
      int r = p >> 4, c = p & 15;
      v8f acc;
#pragma unroll
      for (int i = 0; i < 8; ++i) acc[i] = bias;
#pragma unroll
      for (int tap = 0; tap < 9; ++tap) {     // K = 9 taps x 32 ch = 288
        Frag16 a;
#pragma unroll
        for (int j = 0; j < 8; ++j) {
          int t  = j + g * 8;                 // channel pair index 0..15
          a.u[j] = h32[((r + tap / 3) * 18 + (c + tap % 3)) * 16 + t];
        }
        Frag16 bf;
#pragma unroll
        for (int j = 0; j < 8; ++j) bf.u[j] = sh_w2[tap * 256 + ln * 8 + j];
        acc = wmma_f16(a, bf, acc);
      }
      if (l15 < 4) {                          // only 4 valid output columns
#pragma unroll
        for (int i = 0; i < 8; ++i) {
          int pix = mt * 16 + i + g * 8;
          int pr = pix >> 4, pc = pix & 15;
          size_t idx = (((size_t)n * 4 + l15) * 512 + (r0 + pr)) * 512 + (c0 + pc);
          out[idx] = fmaxf(acc[i], 0.f) * x[idx];
        }
      }
    }
  }
}

// ---------------------------------------------------------------------------
extern "C" void kernel_launch(void* const* d_in, const int* in_sizes, int n_in,
                              void* d_out, int out_size, void* d_ws, size_t ws_size,
                              hipStream_t stream) {
  const float* x     = (const float*)d_in[0];
  const float* w1    = (const float*)d_in[1];
  const float* b1    = (const float*)d_in[2];
  const float* w2    = (const float*)d_in[3];
  const float* b2    = (const float*)d_in[4];
  const float* w_se1 = (const float*)d_in[5];
  const float* b_se1 = (const float*)d_in[6];
  const float* w_se2 = (const float*)d_in[7];
  const float* b_se2 = (const float*)d_in[8];
  float* out = (float*)d_out;

  float* ws_sums  = (float*)d_ws;        // 16*32 floats
  float* ws_scale = ws_sums + 512;       // 16*32 floats

  k_zero<<<1, 512, 0, stream>>>(ws_sums, 512);
  k_conv1_sum<<<dim3(16, 16, 16), 256, 0, stream>>>(x, w1, b1, ws_sums);
  k_se<<<16, 32, 0, stream>>>(ws_sums, w_se1, b_se1, w_se2, b_se2, ws_scale);
  k_fused<<<dim3(32, 32, 16), 256, 0, stream>>>(x, w1, b1, w2, b2, ws_scale, out);
}
